// LIE_12000138625349
// MI455X (gfx1250) — matrix operand order repaired and checked by reading, not compiled, not run on hardware
//
#include <hip/hip_runtime.h>

// Problem constants (reference: B=8, C=256, H=W=56, K=3)
#define BB   8
#define CCH  256
#define HH   56
#define WW   56
#define DD   2304           // C*9
#define XPAD 264            // padded channel stride for the x slice in LDS

typedef __attribute__((ext_vector_type(16))) __bf16 v16bf;
typedef __attribute__((ext_vector_type(8)))  float  v8f;

union Frag16 { v16bf v; uint4 q[2]; };

__device__ __forceinline__ v8f wmma_bf16(v16bf a, v16bf b, v8f c) {
  // (neg_a, A, neg_b, B, c_mod, C, reuse_a, reuse_b)
  return __builtin_amdgcn_wmma_f32_16x16x32_bf16(false, a, false, b, (short)0, c,
                                                 false, false);
}

// A-fragment (16x32 bf16) from a row-major [row][k] buffer with k contiguous.
// ISA layout: lanes 0-15 hold K[0:8]+K[16:24], lanes 16-31 hold K[8:16]+K[24:32].
__device__ __forceinline__ v16bf load_a(const __bf16* base, int stride, int row,
                                        int k0, int lane) {
  const __bf16* p = base + (size_t)row * stride + k0 + ((lane >> 4) << 3);
  Frag16 u;
  u.q[0] = *(const uint4*)(p);
  u.q[1] = *(const uint4*)(p + 16);
  return u.v;
}

// B-fragment (32x16 bf16) from packed weights stored [N][K], K contiguous.
// Lane holds column N = lane%16; lanes 0-15 hold K[0:8]+K[16:24], lanes 16-31 hold K[8:16]+K[24:32].
__device__ __forceinline__ v16bf load_b(const __bf16* w, int nbase, int k0,
                                        int kstride, int lane) {
  const __bf16* p = w + (size_t)(nbase + (lane & 15)) * kstride + k0 +
                    ((lane >> 4) << 3);
  Frag16 u;
  u.q[0] = *(const uint4*)(p);
  u.q[1] = *(const uint4*)(p + 16);
  return u.v;
}

// ---------------------------------------------------------------------------
// Repack weights to bf16 with the K dimension permuted to (tap, channel)
// order: kp = tap*256 + c  <->  original d = c*9 + tap.
//   W1p, W3p : [256][2304]  (n-major, permuted-k contiguous)
//   W2p      : [2304][256]  (permuted-n major, k contiguous)
//   b2p      : [2304] f32   (permuted)
// ---------------------------------------------------------------------------
extern "C" __global__ void __launch_bounds__(256)
lie_repack(const float* __restrict__ W1, const float* __restrict__ W2,
           const float* __restrict__ W3, const float* __restrict__ b2,
           __bf16* __restrict__ W1p, __bf16* __restrict__ W2p,
           __bf16* __restrict__ W3p, float* __restrict__ b2p)
{
  int i = blockIdx.x * 256 + threadIdx.x;
  if (i < DD * CCH) {
    {   // W1p / W3p : [n][kp]
      int n   = i / DD;
      int kp  = i - n * DD;
      int tap = kp >> 8;
      int c   = kp & 255;
      int d   = c * 9 + tap;
      W1p[i] = (__bf16)W1[(size_t)d * CCH + n];
      W3p[i] = (__bf16)W3[(size_t)d * CCH + n];
    }
    {   // W2p : [np][k]
      int np  = i >> 8;
      int k   = i & 255;
      int tap = np >> 8;
      int c   = np & 255;
      int d   = c * 9 + tap;
      W2p[i] = (__bf16)W2[(size_t)k * DD + d];
    }
  }
  if (i < DD) {
    int tap = i >> 8, c = i & 255;
    b2p[i] = b2[c * 9 + tap];
  }
}

// ---------------------------------------------------------------------------
// Fully fused kernel: one workgroup (8 wave32) per (batch image, output row).
//   Phase 0: stage x[b, :, irow-1..irow+1, :] into LDS as bf16 (zero-padded).
//   Phase 1: h1 = relu(t @ W1 + b1)      -> LDS (bf16)
//   Phase 2: loop 72 chunks of 32 d-cols (single barrier, ping-pong a3s):
//              s_chunk = sigmoid(h1 @ W2 + b2); a3 = t*s_chunk -> LDS
//              out    += a3 @ W3[chunk]
//   Epilogue: out + b3 -> global.
// Wave split for GEMM1/3: 2 M-groups x 4 N-quarters; each B fragment is
// fetched from L2 by only 2 waves and feeds two WMMAs (both M-tiles).
// ---------------------------------------------------------------------------
extern "C" __global__ void __launch_bounds__(256)
lie_fused(const float* __restrict__ x,
          const __bf16* __restrict__ W1p, const float* __restrict__ b1,
          const __bf16* __restrict__ W2p, const float* __restrict__ b2p,
          const __bf16* __restrict__ W3p, const float* __restrict__ b3,
          float* __restrict__ out)
{
  __shared__ __bf16 xs[3][64][XPAD];   // 3 src rows x 58(+pad) cols x 256 ch
  __shared__ __bf16 h1s[64][CCH];      // hidden activations
  __shared__ __bf16 a3s[2][64][32];    // (t * s) chunk, ping-pong

  const int tid  = threadIdx.x;
  const int lane = tid & 31;
  const int wave = tid >> 5;
  const int mg   = wave & 1;           // M-group: rows mg*32 .. mg*32+31
  const int nq   = wave >> 1;          // N-quarter: cols nq*64 .. nq*64+63

  const int wg   = blockIdx.x;
  const int bimg = wg / HH;
  const int irow = wg - bimg * HH;

  // ---- Phase 0: stage x slice (coalesced on W; bf16 convert) ----
  for (int idx = tid; idx < 3 * CCH * 64; idx += 256) {
    int cs = idx & 63;                 // LDS col 0..63 (src col cs-1)
    int c  = (idx >> 6) & 255;
    int r  = idx >> 14;                // 0..2 (src row irow+r-1)
    int sr = irow + r - 1;
    int sc = cs - 1;
    float v = 0.0f;
    if (sr >= 0 && sr < HH && sc >= 0 && sc < WW)
      v = x[(((size_t)bimg * CCH + c) * HH + sr) * WW + sc];
    xs[r][cs][c] = (__bf16)v;
  }
  __syncthreads();

  const v8f vzero = {0.f, 0.f, 0.f, 0.f, 0.f, 0.f, 0.f, 0.f};
  v8f acc[8];                          // acc[2*nt + m] : (M-tile m, N-tile nt)
#pragma unroll
  for (int i = 0; i < 8; ++i) acc[i] = vzero;

  const int prow0 = mg * 32 + (lane & 15);
  const int prow1 = prow0 + 16;
  const int pcl0  = prow0 > (WW - 1) ? (WW - 1) : prow0;  // dummy rows clamp
  const int pcl1  = prow1 > (WW - 1) ? (WW - 1) : prow1;

  // ---- Phase 1: GEMM1 (64x2304 @ 2304x256), K permuted as (tap, channel) ----
#pragma unroll 1
  for (int kc = 0; kc < 72; ++kc) {
    int tap = kc >> 3;
    int ki  = tap / 3;
    int kj  = tap - ki * 3;
    int c0  = (kc & 7) << 5;
    v16bf a0 = load_a(&xs[ki][0][0], XPAD, pcl0 + kj, c0, lane);
    v16bf a1 = load_a(&xs[ki][0][0], XPAD, pcl1 + kj, c0, lane);
#pragma unroll
    for (int nt = 0; nt < 4; ++nt) {
      v16bf bf = load_b(W1p, nq * 64 + nt * 16, kc * 32, DD, lane);
      acc[2 * nt + 0] = wmma_bf16(a0, bf, acc[2 * nt + 0]);
      acc[2 * nt + 1] = wmma_bf16(a1, bf, acc[2 * nt + 1]);
    }
  }
  {   // bias + relu -> h1s (bf16)
    int n0 = lane & 15;
    int mr = mg * 32 + ((lane >> 4) << 3);
#pragma unroll
    for (int nt = 0; nt < 4; ++nt) {
      int col    = nq * 64 + nt * 16 + n0;
      float bias = b1[col];
#pragma unroll
      for (int m = 0; m < 2; ++m) {
#pragma unroll
        for (int r = 0; r < 8; ++r) {
          float v = acc[2 * nt + m][r] + bias;
          h1s[mr + m * 16 + r][col] = (__bf16)(v > 0.0f ? v : 0.0f);
        }
      }
    }
  }
  __syncthreads();

  // ---- Phase 2: chunked GEMM2 -> sigmoid*t -> GEMM3 accumulate ----
#pragma unroll
  for (int i = 0; i < 8; ++i) acc[i] = vzero;

  const int sm = wave & 3;             // s-tile row block this wave produces
  const int sn = wave >> 2;            // s-tile col half (16 of 32)

#pragma unroll 1
  for (int kc = 0; kc < 72; ++kc) {
    int buf = kc & 1;
    int tap = kc >> 3;
    int ki  = tap / 3;
    int kj  = tap - ki * 3;
    int c0  = (kc & 7) << 5;

    // produce: s tile = sigmoid(h1 @ W2p[:, chunk] + b2p), times t
    v8f sacc = vzero;
#pragma unroll
    for (int kk = 0; kk < 8; ++kk) {
      v16bf a  = load_a(&h1s[0][0], CCH, sm * 16 + (lane & 15), kk * 32, lane);
      v16bf bf = load_b(W2p, kc * 32 + sn * 16, kk * 32, CCH, lane);
      sacc = wmma_bf16(a, bf, sacc);
    }
    {
      int n0     = lane & 15;
      int npcol  = kc * 32 + sn * 16 + n0;
      float bias = b2p[npcol];
      int cch    = c0 + sn * 16 + n0;      // channel within this tap
      int mrow   = sm * 16 + ((lane >> 4) << 3);
#pragma unroll
      for (int r = 0; r < 8; ++r) {
        int p  = mrow + r;
        int pc = p > (WW - 1) ? (WW - 1) : p;
        float tv = (float)xs[ki][pc + kj][cch];
        float sv = sacc[r] + bias;
        sv = 1.0f / (1.0f + __expf(-sv));
        a3s[buf][p][sn * 16 + n0] = (__bf16)(tv * sv);
      }
    }
    // One barrier per chunk: produce(kc) visible; consume(kc-1) of the other
    // buffer is transitively ordered before produce(kc+1) by this barrier.
    __syncthreads();

    // consume: out += a3 @ W3p[chunk, :]
    v16bf a30 = load_a(&a3s[buf][0][0], 32, mg * 32 + (lane & 15), 0, lane);
    v16bf a31 = load_a(&a3s[buf][0][0], 32, mg * 32 + 16 + (lane & 15), 0, lane);
#pragma unroll
    for (int nt = 0; nt < 4; ++nt) {
      v16bf bf = load_b(W3p, nq * 64 + nt * 16, kc * 32, DD, lane);
      acc[2 * nt + 0] = wmma_bf16(a30, bf, acc[2 * nt + 0]);
      acc[2 * nt + 1] = wmma_bf16(a31, bf, acc[2 * nt + 1]);
    }
  }

  // ---- Epilogue: + b3, store fp32 output ----
  {
    int n0    = lane & 15;
    int mr    = mg * 32 + ((lane >> 4) << 3);
    size_t lb = (size_t)bimg * (HH * WW) + (size_t)irow * WW;
#pragma unroll
    for (int nt = 0; nt < 4; ++nt) {
      int col    = nq * 64 + nt * 16 + n0;
      float bias = b3[col];
#pragma unroll
      for (int m = 0; m < 2; ++m) {
#pragma unroll
        for (int r = 0; r < 8; ++r) {
          int p = mr + m * 16 + r;
          if (p < WW)
            out[(lb + p) * CCH + col] = acc[2 * nt + m][r] + bias;
        }
      }
    }
  }
}

// ---------------------------------------------------------------------------
extern "C" void kernel_launch(void* const* d_in, const int* in_sizes, int n_in,
                              void* d_out, int out_size, void* d_ws, size_t ws_size,
                              hipStream_t stream) {
  const float* x  = (const float*)d_in[0];
  const float* W1 = (const float*)d_in[1];
  const float* b1 = (const float*)d_in[2];
  const float* W2 = (const float*)d_in[3];
  const float* b2 = (const float*)d_in[4];
  const float* W3 = (const float*)d_in[5];
  const float* b3 = (const float*)d_in[6];

  char* ws = (char*)d_ws;
  const size_t wbytes = (size_t)DD * CCH * sizeof(__bf16);   // 1,179,648 B each
  __bf16* W1p = (__bf16*)(ws);
  __bf16* W2p = (__bf16*)(ws + wbytes);
  __bf16* W3p = (__bf16*)(ws + 2 * wbytes);
  float*  b2p = (float*)(ws + 3 * wbytes);                   // total ~3.55 MB

  int n = DD * CCH;
  lie_repack<<<(n + 255) / 256, 256, 0, stream>>>(W1, W2, W3, b2,
                                                  W1p, W2p, W3p, b2p);
  lie_fused<<<BB * HH, 256, 0, stream>>>(x, W1p, b1, W2p, b2p, W3p, b3,
                                         (float*)d_out);
}
